// DKF_37271726195451
// MI455X (gfx1250) — matrix operand order repaired and checked by reading, not compiled, not run on hardware
//
#include <hip/hip_runtime.h>
#include <hip/hip_bf16.h>
#include <cstddef>

// ---------------------------------------------------------------------------
// DKF scan for MI455X (gfx1250, wave32).
// One wave owns a 16-row batch tile and executes the entire T=1000 recurrence.
// All GEMMs lowered to v_wmma_f32_16x16x32_f16 (f16 A/B, f32 accumulate).
// Weights live in VGPRs as prebuilt B-fragments. LDS tiles are zero-padded to
// K=32 so A-fragment builds are unconditional (no divergent guarded ds loads).
// 32 blocks x 32 threads; next-step xs hidden via global_prefetch_b8.
// ---------------------------------------------------------------------------

typedef __attribute__((ext_vector_type(16))) _Float16 v16h;
typedef __attribute__((ext_vector_type(8)))  float    v8f;

#define LATENT_D 10
#define GT_SIZE  64
#define EZ_SIZE  32
#define NW       10
#define BATCH    512
#define TSTEPS   1000

static __device__ __forceinline__ v8f zero8() {
  v8f z = {0.f,0.f,0.f,0.f,0.f,0.f,0.f,0.f};
  return z;
}
static __device__ __forceinline__ v8f wmma16(v16h a, v16h b, v8f c) {
  // 8 args: (neg_a, A, neg_b, B, c_mod, C, reuse_a, reuse_b)
  return __builtin_amdgcn_wmma_f32_16x16x32_f16(false, a, false, b, (short)0, c,
                                                false, false);
}
static __device__ __forceinline__ float rcp_fast(float x) {
  return __builtin_amdgcn_rcpf(x);            // v_rcp_f32 (TRANS)
}
static __device__ __forceinline__ float fast_tanh(float x) {
  float cx = fminf(fmaxf(x, -15.f), 15.f);
  float e  = __expf(2.f * cx);                // v_exp_f32 (TRANS)
  return (e - 1.f) * rcp_fast(e + 1.f);
}
static __device__ __forceinline__ float softplusf(float x) {
  return (x > 15.f) ? x : __logf(1.f + __expf(x));
}
static __device__ __forceinline__ float sigmoidf_(float x) {
  return rcp_fast(1.f + __expf(-x));
}

// B fragment (K x N, f16): N = lane&15; lanes 0-15: K={0..7,16..23},
// lanes 16-31: K={8..15,24..31} (within 32-row K-chunk).
// Built once before the T loop; guards here are loop-invariant and cheap.
static __device__ __forceinline__ v16h b_frag(const float* W, int fan_in, int fan_out,
                                              int kc, int nt, int lane) {
  int n = nt * 16 + (lane & 15);
  int kb = kc * 32 + ((lane & 16) ? 8 : 0);
  v16h f;
#pragma unroll
  for (int j = 0; j < 16; ++j) {
    int k = kb + (j < 8 ? j : j + 8);
    float v = (k < fan_in && n < fan_out) ? W[k * fan_out + n] : 0.f;
    f[j] = (_Float16)v;
  }
  return f;
}

// A fragment (16 x 32, f16) from an f32 LDS tile with leading dim `ld`.
// UNCONDITIONAL reads: callers guarantee buf[m*ld + k] is valid (zero-padded)
// for all k in [kc*32, kc*32+32). lanes 0-15: M=lane, K={0..7,16..23};
// lanes 16-31: M=lane-16, K={8..15,24..31}.
static __device__ __forceinline__ v16h a_frag(const float* buf, int ld, int kc,
                                              int lane) {
  int m  = lane & 15;
  int kb = kc * 32 + ((lane & 16) ? 8 : 0);
  const float* row = buf + m * ld + kb;
  v16h f;
#pragma unroll
  for (int j = 0; j < 16; ++j) {
    int k = (j < 8 ? j : j + 8);
    f[j] = (_Float16)row[k];
  }
  return f;
}

__global__ __launch_bounds__(32)
void dkf_scan_kernel(const float* __restrict__ xs, const float* __restrict__ eps,
                     const unsigned char* __restrict__ mask,
                     const float* __restrict__ Wgt, const float* __restrict__ bgt,
                     const float* __restrict__ Wez, const float* __restrict__ bez,
                     const float* __restrict__ Wmu, const float* __restrict__ bmu,
                     const float* __restrict__ Wsg, const float* __restrict__ bsg,
                     const float* __restrict__ Wn1, const float* __restrict__ bn1,
                     const float* __restrict__ Wn2, const float* __restrict__ bn2,
                     const float* __restrict__ Wm1, const float* __restrict__ bm1,
                     const float* __restrict__ Wm2, const float* __restrict__ bm2,
                     const float* __restrict__ Wml, const float* __restrict__ bml,
                     const float* __restrict__ Wdz, const float* __restrict__ bdz,
                     float* __restrict__ z_out, float* __restrict__ kl_out,
                     float* __restrict__ iw_out) {
  __shared__ float xbuf[16 * GT_SIZE];   // xs tile, overwritten in place with xt
  __shared__ float hbuf[16 * EZ_SIZE];   // ez activations (K=32, no pad needed)
  __shared__ float sbuf[16 * 32];        // recurrent sample, cols 10..31 == 0
  __shared__ float tA[16 * 32];          // prior-net staging, cols 10..31 == 0
  __shared__ float tB[16 * 32];          // prior-net staging, cols 10..31 == 0
  __shared__ float ebuf[16 * 16];        // eps tile, cols 10..15 == 0
  __shared__ float iwbuf[16 * 16];       // iwae terms for N-reduction
  __shared__ float zbuf[16 * 16];        // staged z outputs (coalesced store)
  __shared__ float klbuf[16 * 16];       // staged kl outputs
  __shared__ float mbuf[16];             // per-row mask

  const int lane = threadIdx.x;          // wave32
  const int n    = lane & 15;            // C/D column index
  const int mh   = (lane & 16) ? 8 : 0;  // C/D row half
  const int m0   = blockIdx.x * 16;      // first batch row of this tile

  // ---- Preload all weights as f16 B-fragments (held in VGPRs for all T) ----
  v16h Bgt[4];
#pragma unroll
  for (int t = 0; t < 4; ++t) Bgt[t] = b_frag(Wgt, LATENT_D, GT_SIZE, 0, t, lane);
  v16h Bez[2][2];
#pragma unroll
  for (int kc = 0; kc < 2; ++kc)
#pragma unroll
    for (int t = 0; t < 2; ++t) Bez[kc][t] = b_frag(Wez, GT_SIZE, EZ_SIZE, kc, t, lane);
  v16h Bmu = b_frag(Wmu, EZ_SIZE, LATENT_D, 0, 0, lane);
  v16h Bsg = b_frag(Wsg, EZ_SIZE, LATENT_D, 0, 0, lane);
  v16h Bn1 = b_frag(Wn1, LATENT_D, NW,       0, 0, lane);
  v16h Bn2 = b_frag(Wn2, NW,       LATENT_D, 0, 0, lane);
  v16h Bm1 = b_frag(Wm1, LATENT_D, NW,       0, 0, lane);
  v16h Bm2 = b_frag(Wm2, NW,       LATENT_D, 0, 0, lane);
  v16h Bml = b_frag(Wml, LATENT_D, LATENT_D, 0, 0, lane);
  v16h Bdz = b_frag(Wdz, NW,       LATENT_D, 0, 0, lane);

  // Per-lane (per-column) biases; zero in pad columns so pad math yields 0.
  float bias_gt[4];
#pragma unroll
  for (int t = 0; t < 4; ++t) bias_gt[t] = bgt[t * 16 + n];
  float bias_ez[2];
#pragma unroll
  for (int t = 0; t < 2; ++t) bias_ez[t] = bez[t * 16 + n];
  const float bias_mu = (n < LATENT_D) ? bmu[n] : 0.f;
  const float bias_sg = (n < LATENT_D) ? bsg[n] : 0.f;
  const float bias_n1 = (n < NW)       ? bn1[n] : 0.f;
  const float bias_n2 = (n < LATENT_D) ? bn2[n] : 0.f;
  const float bias_m1 = (n < NW)       ? bm1[n] : 0.f;
  const float bias_m2 = (n < LATENT_D) ? bm2[n] : 0.f;
  const float bias_ml = (n < LATENT_D) ? bml[n] : 0.f;
  const float bias_dz = (n < LATENT_D) ? bdz[n] : 0.f;

  // init: zero sample + all pad regions once.
  for (int i = lane; i < 16 * 32; i += 32) { sbuf[i] = 0.f; tA[i] = 0.f; tB[i] = 0.f; }
  for (int i = lane; i < 256; i += 32) ebuf[i] = 0.f;
  __syncthreads();

  for (int t = 0; t < TSTEPS; ++t) {
    // -------- Phase A: stage xs / eps / mask into LDS (coalesced) ----------
    {
      const int r  = lane & 15;
      const int c0 = (lane & 16) ? 32 : 0;
      const float* src = xs + ((size_t)(m0 + r) * TSTEPS + t) * GT_SIZE + c0;
#pragma unroll
      for (int j = 0; j < 8; ++j) {
        float4 v = *reinterpret_cast<const float4*>(src + 4 * j);
        *reinterpret_cast<float4*>(&xbuf[r * GT_SIZE + c0 + 4 * j]) = v;
      }
      if (t + 1 < TSTEPS) {
        __builtin_prefetch(src + GT_SIZE, 0, 1);   // -> global_prefetch_b8
      }
      if (lane < 16) {
        const float* esrc = eps + ((size_t)(m0 + lane) * TSTEPS + t) * LATENT_D;
#pragma unroll
        for (int j = 0; j < 5; ++j) {
          float2 v = *reinterpret_cast<const float2*>(esrc + 2 * j);
          *reinterpret_cast<float2*>(&ebuf[lane * 16 + 2 * j]) = v;
        }
        mbuf[lane] = mask[(size_t)(m0 + lane) * TSTEPS + t] ? 1.f : 0.f;
      }
    }
    __syncthreads();

    // -------- Phase B: xt = 0.5*(tanh(sample@Wgt + b) + xs) ----------------
    v16h As = a_frag(sbuf, 32, 0, lane);
#pragma unroll
    for (int nt = 0; nt < 4; ++nt) {
      v8f acc = wmma16(As, Bgt[nt], zero8());
      int col = nt * 16 + n;
#pragma unroll
      for (int v = 0; v < 8; ++v) {
        int m = v + mh;
        float val = 0.5f * (fast_tanh(acc[v] + bias_gt[nt]) + xbuf[m * GT_SIZE + col]);
        xbuf[m * GT_SIZE + col] = val;    // in-place: each (m,col) owned by one lane
      }
    }
    __syncthreads();

    // -------- Phase C: h = tanh(xt@Wez + b) --------------------------------
    v16h Ax0 = a_frag(xbuf, GT_SIZE, 0, lane);
    v16h Ax1 = a_frag(xbuf, GT_SIZE, 1, lane);
#pragma unroll
    for (int nt = 0; nt < 2; ++nt) {
      v8f c = wmma16(Ax0, Bez[0][nt], zero8());
      c = wmma16(Ax1, Bez[1][nt], c);
      int col = nt * 16 + n;
#pragma unroll
      for (int v = 0; v < 8; ++v)
        hbuf[(v + mh) * EZ_SIZE + col] = fast_tanh(c[v] + bias_ez[nt]);
    }
    __syncthreads();

    // -------- Phase D: posterior heads -------------------------------------
    v16h Ah  = a_frag(hbuf, EZ_SIZE, 0, lane);
    v8f qmu_t = wmma16(Ah, Bmu, zero8());
    v8f qsg_t = wmma16(Ah, Bsg, zero8());

    // -------- Phase E: prior network (NonlinearDZ) -------------------------
    v8f a1_t = wmma16(As, Bn1, zero8());
    v8f m1_t = wmma16(As, Bm1, zero8());
    v8f ml_t = wmma16(As, Bml, zero8());
#pragma unroll
    for (int v = 0; v < 8; ++v) {
      int m = v + mh;
      // Pad columns (n>=10) receive exact 0 (zero weights + zero bias + relu).
      tA[m * 32 + n] = fmaxf(a1_t[v] + bias_n1, 0.f);   // relu(nu1)
      tB[m * 32 + n] = fmaxf(m1_t[v] + bias_m1, 0.f);   // relu(mun1)
    }
    __syncthreads();
    v16h Aa1 = a_frag(tA, 32, 0, lane);
    v16h Am1 = a_frag(tB, 32, 0, lane);
    v8f nupre_t = wmma16(Aa1, Bn2, zero8());
    v8f mun_t   = wmma16(Am1, Bm2, zero8());
    __syncthreads();                                    // tA consumed by all lanes
#pragma unroll
    for (int v = 0; v < 8; ++v)
      tA[(v + mh) * 32 + n] = fmaxf(mun_t[v] + bias_m2, 0.f);  // relu(mun)
    __syncthreads();
    v16h Amn = a_frag(tA, 32, 0, lane);
    v8f dz_t = wmma16(Amn, Bdz, zero8());

    // -------- Phase F: KL / sample update / IWAE (elementwise, unguarded) --
#pragma unroll
    for (int v = 0; v < 8; ++v) {
      int m = v + mh;
      float qm = qmu_t[v] + bias_mu;
      float qs = softplusf(qsg_t[v] + bias_sg);
      float nu = sigmoidf_(nupre_t[v] + bias_n2);
      float pl = ml_t[v] + bias_ml;
      float pn = mun_t[v] + bias_m2;
      float pm = (1.f - nu) * pl + nu * pn;
      float ps = softplusf(dz_t[v] + bias_dz);
      float e  = ebuf[m * 16 + n];
      float mk = mbuf[m];

      float rqs = rcp_fast(qs), rps = rcp_fast(ps);
      float lqs = __logf(qs),   lps = __logf(ps);

      float dmu = qm - pm;
      float kl = (lps - lqs) + (qs * qs + dmu * dmu) * (0.5f * rps * rps) - 0.5f;

      float nw = qs * e + qm;
      float pz = fminf(fmaxf(ps * e + pm, -1e6f), 1e6f);
      nw = (mk != 0.f) ? nw : pz;

      float dq = (nw - qm) * rqs;
      float dp = (nw - pm) * rps;
      float term = 0.5f * (dp * dp - dq * dq) + (lps - lqs);

      // Pad columns produce exact zeros here (all inputs zero), so no guards.
      iwbuf[m * 16 + n] = term;
      zbuf [m * 16 + n] = nw;
      klbuf[m * 16 + n] = kl;
      sbuf [m * 32 + n] = nw;   // carry for step t+1 (cols 16..31 stay 0)
    }
    __syncthreads();

    // -------- Phase G: coalesced outputs + iwae reduction ------------------
    if (lane < 16) {
      size_t base = ((size_t)(m0 + lane) * TSTEPS + t) * LATENT_D;  // even -> 8B aligned
#pragma unroll
      for (int j = 0; j < 5; ++j) {
        float2 zv  = *reinterpret_cast<const float2*>(&zbuf [lane * 16 + 2 * j]);
        float2 kv  = *reinterpret_cast<const float2*>(&klbuf[lane * 16 + 2 * j]);
        *reinterpret_cast<float2*>(z_out  + base + 2 * j) = zv;
        *reinterpret_cast<float2*>(kl_out + base + 2 * j) = kv;
      }
      float s = 0.f;
#pragma unroll
      for (int d = 0; d < LATENT_D; ++d) s += iwbuf[lane * 16 + d];
      iw_out[(size_t)(m0 + lane) * TSTEPS + t] = s;
    }
    __syncthreads();  // sbuf/iwbuf stable before next iteration
  }
}

extern "C" void kernel_launch(void* const* d_in, const int* in_sizes, int n_in,
                              void* d_out, int out_size, void* d_ws, size_t ws_size,
                              hipStream_t stream) {
  (void)in_sizes; (void)n_in; (void)out_size; (void)d_ws; (void)ws_size;
  const float*         xs   = (const float*)d_in[0];
  const float*         eps  = (const float*)d_in[1];
  const unsigned char* mask = (const unsigned char*)d_in[2];
  const float* Wgt = (const float*)d_in[3];  const float* bgt = (const float*)d_in[4];
  const float* Wez = (const float*)d_in[5];  const float* bez = (const float*)d_in[6];
  const float* Wmu = (const float*)d_in[7];  const float* bmu = (const float*)d_in[8];
  const float* Wsg = (const float*)d_in[9];  const float* bsg = (const float*)d_in[10];
  const float* Wn1 = (const float*)d_in[11]; const float* bn1 = (const float*)d_in[12];
  const float* Wn2 = (const float*)d_in[13]; const float* bn2 = (const float*)d_in[14];
  const float* Wm1 = (const float*)d_in[15]; const float* bm1 = (const float*)d_in[16];
  const float* Wm2 = (const float*)d_in[17]; const float* bm2 = (const float*)d_in[18];
  const float* Wml = (const float*)d_in[19]; const float* bml = (const float*)d_in[20];
  const float* Wdz = (const float*)d_in[21]; const float* bdz = (const float*)d_in[22];

  float* z  = (float*)d_out;
  float* kl = z  + (size_t)BATCH * TSTEPS * LATENT_D;
  float* iw = kl + (size_t)BATCH * TSTEPS * LATENT_D;

  dim3 grid(BATCH / 16), block(32);
  dkf_scan_kernel<<<grid, block, 0, stream>>>(xs, eps, mask,
                                              Wgt, bgt, Wez, bez, Wmu, bmu, Wsg, bsg,
                                              Wn1, bn1, Wn2, bn2, Wm1, bm1, Wm2, bm2,
                                              Wml, bml, Wdz, bdz,
                                              z, kl, iw);
}
